// BiDirectionLSTM_24764781428895
// MI455X (gfx1250) — compile-verified, matrix operand order checked by reading, not run on hardware
//
#include <hip/hip_runtime.h>
#include <hip/hip_bf16.h>
#include <math.h>

typedef unsigned short u16;
typedef unsigned int   u32;

#define B_   32
#define T_   512
#define D_   512
#define H_   512
#define N4H  2048
#define M_   (B_ * T_)          // 16384

typedef __attribute__((ext_vector_type(16))) __bf16 v16bf;
typedef __attribute__((ext_vector_type(8)))  float  v8f;
typedef int v2i __attribute__((vector_size(8)));   // matches async-builtin param type

// ---- CDNA5 async global->LDS path (guarded; fallback = sync copy) ----
#if defined(__has_builtin)
# if __has_builtin(__builtin_amdgcn_global_load_async_to_lds_b64)
#  define HAVE_ASYNC 1
# endif
#endif
#ifndef HAVE_ASYNC
# define HAVE_ASYNC 0
#endif

__device__ __forceinline__ void async_copy_b64(const void* g, void* l) {
#if HAVE_ASYNC
    __builtin_amdgcn_global_load_async_to_lds_b64(
        (__attribute__((address_space(1))) v2i*)(uintptr_t)g,
        (__attribute__((address_space(3))) v2i*)(u32)(uintptr_t)l,
        0, 0);
#else
    *(uint2*)l = *(const uint2*)g;
#endif
}
__device__ __forceinline__ void wait_async0() {
#if HAVE_ASYNC
# if __has_builtin(__builtin_amdgcn_s_wait_asynccnt)
    __builtin_amdgcn_s_wait_asynccnt(0);
# else
    asm volatile("s_wait_asynccnt 0x0" ::: "memory");
# endif
#endif
}

// ---------- helpers ----------
__device__ __forceinline__ u16 f2bf(float f) {
    u32 u = __float_as_uint(f);
    u32 r = (u + 0x7FFFu + ((u >> 16) & 1u)) >> 16;  // RNE
    return (u16)r;
}
__device__ __forceinline__ float bf2f(u16 h) {
    return __uint_as_float(((u32)h) << 16);
}
__device__ __forceinline__ float hsig(float x) {
    return fminf(fmaxf(0.2f * x + 0.5f, 0.0f), 1.0f);
}
__device__ __forceinline__ v16bf load_frag(const u16* p) {
    union { uint4 q[2]; v16bf v; } u;
    const uint4* q = (const uint4*)p;
    u.q[0] = q[0]; u.q[1] = q[1];
    return u.v;
}
__device__ __forceinline__ v16bf load_frag2(const u16* p0, const u16* p1) {
    union { uint4 q[2]; v16bf v; } u;
    u.q[0] = *(const uint4*)p0; u.q[1] = *(const uint4*)p1;
    return u.v;
}

// ---------- init / conversion kernels ----------
__global__ void zero_f32(float* p, int n) {
    for (int i = blockIdx.x * blockDim.x + threadIdx.x; i < n; i += gridDim.x * blockDim.x)
        p[i] = 0.0f;
}
__global__ void zero_u32(u32* p, int n) {
    for (int i = blockIdx.x * blockDim.x + threadIdx.x; i < n; i += gridDim.x * blockDim.x)
        p[i] = 0u;
}
__global__ void cvt_bf16(const float* s, u16* d, int n) {
    for (int i = blockIdx.x * blockDim.x + threadIdx.x; i < n; i += gridDim.x * blockDim.x)
        d[i] = f2bf(s[i]);
}

// Pack 4 gate matrices [512 x 512] (row-major, K x N) into WMMA B-fragment order:
// frag f = (g*512 + ntl*16 + kt); within frag: lane L (32) x elem e (16) bf16.
__global__ void pack_frag4(const float* s0, const float* s1, const float* s2,
                           const float* s3, u16* dst) {
    int id = blockIdx.x * blockDim.x + threadIdx.x;
    if (id >= 4 * 262144) return;
    int g   = id >> 18;
    int rem = id & 262143;
    int ntl = rem >> 13;          // 0..31
    int kt  = (rem >> 9) & 15;    // 0..15
    int L   = (rem >> 4) & 31;    // lane
    int e   = rem & 15;           // element
    const float* s = (g == 0) ? s0 : (g == 1) ? s1 : (g == 2) ? s2 : s3;
    int k = kt * 32 + e + ((e >= 8) ? 8 : 0) + ((L >= 16) ? 8 : 0);
    int n = ntl * 16 + (L & 15);
    dst[id] = f2bf(s[k * H_ + n]);
}

__global__ void bias_cat(const float* a0, const float* a1, const float* a2, const float* a3,
                         const float* a4, const float* a5, const float* a6, const float* a7,
                         float* bc) {
    int id = blockIdx.x * blockDim.x + threadIdx.x;
    if (id >= 2 * N4H) return;
    int dir = id >> 11, n = id & 2047, g = n >> 9, h = n & 511;
    const float* bp[8] = {a0, a1, a2, a3, a4, a5, a6, a7};
    bc[id] = bp[dir * 4 + g][h];
}

// ---------- Phase 1: input projections  P[dir][m=t*32+b][n=g*512+h] = x @ Wcat + b ----------
// Double-buffered A staging: async global->LDS copies for K-slice kk+1 overlap the
// WMMA stream on K-slice kk; ASYNCcnt waited only at the top of each iteration.
__global__ __launch_bounds__(256)
void gemm_in(const u16* __restrict__ xb, const u16* __restrict__ Wpk,
             const float* __restrict__ bcat, u16* __restrict__ P) {
    const int Mt = blockIdx.x, Nt = blockIdx.y, dir = blockIdx.z;
    const u16*  Wd = Wpk + (size_t)dir * (2048u * 512u);
    u16*        Pd = P   + (size_t)dir * ((size_t)M_ * N4H);
    const float* bd = bcat + dir * N4H;

    __shared__ __align__(16) u16 As[2][128][40];   // double-buffered 128x32 + pad

    const int tid = threadIdx.x;
    const int w   = tid >> 5, L = tid & 31;
    const int wm  = w & 1, wn = w >> 1;            // 2 x 4 wave grid; wave tile 64M x 32N
    const int rlo = L & 15;
    const int e0  = (L < 16) ? 0 : 8;

    // staging indices (constant per thread)
    const int sr = tid >> 3, sc = (tid & 7) * 4;   // 32 rows per pass, 8B per thread

    v8f acc[4][2];
    const v8f vz = {0.f, 0.f, 0.f, 0.f, 0.f, 0.f, 0.f, 0.f};
    for (int mt = 0; mt < 4; ++mt)
        for (int j = 0; j < 2; ++j) acc[mt][j] = vz;

#define STAGE(kk)                                                              \
    do {                                                                       \
        int buf_ = (kk) & 1;                                                   \
        for (int p_ = 0; p_ < 4; ++p_) {                                       \
            int r_ = p_ * 32 + sr;                                             \
            int m_ = Mt * 128 + r_;                                            \
            int t_ = m_ >> 5, b_ = m_ & 31;                                    \
            int tt_ = dir ? (T_ - 1 - t_) : t_;                                \
            const u16* src_ = xb + ((size_t)(b_ * T_ + tt_)) * D_ +            \
                              (kk) * 32 + sc;                                  \
            async_copy_b64(src_, &As[buf_][r_][sc]);                           \
        }                                                                      \
    } while (0)

    STAGE(0);
    for (int kk = 0; kk < 16; ++kk) {
        wait_async0();
        __syncthreads();
        if (kk + 1 < 16) STAGE(kk + 1);

        const int buf = kk & 1;
        v16bf a[4];
        for (int mt = 0; mt < 4; ++mt) {
            const u16* ap = &As[buf][wm * 64 + mt * 16 + rlo][e0];
            a[mt] = load_frag2(ap, ap + 16);
        }
        for (int j = 0; j < 2; ++j) {
            int ntg = Nt * 8 + wn * 2 + j;
            const u16* bp = Wd + ((size_t)(ntg * 16 + kk)) * 512 + L * 16;
            v16bf bfr = load_frag(bp);
            for (int mt = 0; mt < 4; ++mt)
                acc[mt][j] = __builtin_amdgcn_wmma_f32_16x16x32_bf16(
                    false, a[mt], false, bfr, (short)0, acc[mt][j], false, false);
        }
        __syncthreads();   // keep LDS reads of buf complete before it is re-staged
    }
#undef STAGE

    // epilogue: +bias, cast bf16, store
    const int hiadd = (L >= 16) ? 8 : 0;
    for (int mt = 0; mt < 4; ++mt)
        for (int j = 0; j < 2; ++j)
            for (int r = 0; r < 8; ++r) {
                int m = Mt * 128 + wm * 64 + mt * 16 + r + hiadd;
                int n = Nt * 128 + wn * 32 + j * 16 + rlo;
                Pd[(size_t)m * N4H + n] = f2bf(acc[mt][j][r] + bd[n]);
            }
}

// ---------- Phase 2: recurrent scan (persistent, per-step device-wide barrier) ----------
// grid (16 chunks, 2 dirs), block 128 (4 waves). Wave w -> C-tile (mt=w&1, nt=w>>1)
// of this WG's 32-wide hidden chunk; all 4 gates of that tile live in one wave so
// the cell state c stays in registers for the whole sequence. Preactivations for
// step t are fetched BEFORE the K-loop so their latency hides behind 64 WMMAs.
__global__ __launch_bounds__(128)
void lstm_scan(const u16* __restrict__ Upk, const u16* __restrict__ P,
               u16* __restrict__ hbuf, float* __restrict__ out, u32* __restrict__ cnt) {
    const int chunk = blockIdx.x, dir = blockIdx.y;
    const int tid = threadIdx.x, w = tid >> 5, L = tid & 31;
    const int mt = w & 1, nt = w >> 1;
    const int rlo = L & 15;
    const int hiadd = (L >= 16) ? 8 : 0;
    const int e0 = (L < 16) ? 0 : 8;

    const u16* Ud = Upk + (size_t)dir * (2048u * 512u);
    const u16* Pd = P   + (size_t)dir * ((size_t)M_ * N4H);
    u16* hcur  = hbuf + (size_t)(dir * 2 + 0) * (B_ * H_);
    u16* hnext = hbuf + (size_t)(dir * 2 + 1) * (B_ * H_);
    u32* c = cnt + dir;

    const int hc = chunk * 32 + nt * 16 + rlo;     // hidden column of this lane
    // loop-invariant U fragment bases (4 gates)
    const u16* ub[4];
    for (int g = 0; g < 4; ++g)
        ub[g] = Ud + ((size_t)((g * 32 + chunk * 2 + nt) * 16)) * 512 + L * 16;
    const int arow = (mt * 16 + rlo) * H_ + e0;    // A-frag base within h buffer

    float cst[8];
    for (int r = 0; r < 8; ++r) cst[r] = 0.0f;

    const v8f vz = {0.f, 0.f, 0.f, 0.f, 0.f, 0.f, 0.f, 0.f};

    for (int t = 0; t < T_; ++t) {
        // ---- issue preactivation loads early (independent of h) ----
        float pre[4][8];
        {
            const u16* pt = Pd + (size_t)(t * 32) * N4H + hc;
            for (int r = 0; r < 8; ++r) {
                int b = mt * 16 + r + hiadd;
                const u16* pr = pt + (size_t)b * N4H;
                pre[0][r] = bf2f(pr[0 * H_]);
                pre[1][r] = bf2f(pr[1 * H_]);
                pre[2][r] = bf2f(pr[2 * H_]);
                pre[3][r] = bf2f(pr[3 * H_]);
            }
        }
        if (t + 1 < T_)
            __builtin_prefetch(Pd + (size_t)((t + 1) * 32 + mt * 16 + rlo) * N4H, 0, 1);

        // ---- h @ U for this tile, all 4 gates ----
        v8f acc[4] = {vz, vz, vz, vz};
        for (int kt = 0; kt < 16; ++kt) {
            const u16* ap = hcur + arow + kt * 32;
            v16bf a = load_frag2(ap, ap + 16);
            for (int g = 0; g < 4; ++g) {
                v16bf bfr = load_frag(ub[g] + (size_t)kt * 512);
                acc[g] = __builtin_amdgcn_wmma_f32_16x16x32_bf16(
                    false, a, false, bfr, (short)0, acc[g], false, false);
            }
        }

        // ---- LSTM cell ----
        for (int r = 0; r < 8; ++r) {
            int b = mt * 16 + r + hiadd;
            float i_t = hsig(acc[0][r] + pre[0][r]);
            float f_t = hsig(acc[1][r] + pre[1][r]);
            float g_t = tanhf(acc[2][r] + pre[2][r]);
            float o_t = hsig(acc[3][r] + pre[3][r]);
            float cv  = f_t * cst[r] + i_t * g_t;
            cst[r] = cv;
            float hv = o_t * tanhf(cv);
            hnext[b * H_ + hc] = f2bf(hv);
            atomicAdd(&out[((size_t)b * T_ + t) * H_ + hc], hv);  // fwd+bwd sum
        }

        // ---- per-direction step barrier (monotonic counter; graph-replay safe) ----
        __threadfence();
        __syncthreads();
        if (tid == 0) {
            atomicAdd(c, 1u);
            u32 target = (u32)(t + 1) * 16u;
            while (__hip_atomic_load(c, __ATOMIC_RELAXED, __HIP_MEMORY_SCOPE_AGENT) < target)
                __builtin_amdgcn_s_sleep(2);
        }
        __syncthreads();
        __threadfence();
        u16* tmp = hnext; hnext = hcur; hcur = tmp;
    }
}

// ---------- host launcher ----------
extern "C" void kernel_launch(void* const* d_in, const int* in_sizes, int n_in,
                              void* d_out, int out_size, void* d_ws, size_t ws_size,
                              hipStream_t stream) {
    (void)in_sizes; (void)n_in; (void)out_size; (void)ws_size;
    // inputs (setup_inputs dict order):
    // 0: x | 1..4: W_i,W_f,W_c,W_o | 5..8: Wb_* | 9..12: U_* | 13..16: Ub_* |
    // 17..20: b_* | 21..24: bb_*
    const float* x = (const float*)d_in[0];

    char* wsb = (char*)d_ws;
    u16*   xb   = (u16*)(wsb + 0);                         // 16 MB  : x in bf16
    u16*   Wpk  = (u16*)(wsb + 16777216);                  // 4 MB   : packed W (2 dirs)
    u16*   Upk  = (u16*)(wsb + 20971520);                  // 4 MB   : packed U (2 dirs)
    float* bcat = (float*)(wsb + 25165824);                // 16 KB  : biases [2][2048]
    u16*   P    = (u16*)(wsb + 25182208);                  // 128 MB : gate preacts bf16
    u16*   hbuf = (u16*)(wsb + 159399936);                 // 128 KB : h double buffers
    u32*   cnt  = (u32*)(wsb + 159531008);                 // counters

    float* out = (float*)d_out;

    // 1) zero output, h buffers, counters
    zero_f32<<<2048, 256, 0, stream>>>(out, B_ * T_ * H_);
    zero_u32<<<129, 256, 0, stream>>>((u32*)hbuf, (131072 + 256) / 4);

    // 2) convert x -> bf16
    cvt_bf16<<<4096, 256, 0, stream>>>(x, xb, B_ * T_ * D_);

    // 3) pack weights into WMMA fragment order (per direction)
    for (int dir = 0; dir < 2; ++dir) {
        pack_frag4<<<4096, 256, 0, stream>>>(
            (const float*)d_in[1 + dir * 4 + 0], (const float*)d_in[1 + dir * 4 + 1],
            (const float*)d_in[1 + dir * 4 + 2], (const float*)d_in[1 + dir * 4 + 3],
            Wpk + (size_t)dir * (2048u * 512u));
        pack_frag4<<<4096, 256, 0, stream>>>(
            (const float*)d_in[9 + dir * 4 + 0], (const float*)d_in[9 + dir * 4 + 1],
            (const float*)d_in[9 + dir * 4 + 2], (const float*)d_in[9 + dir * 4 + 3],
            Upk + (size_t)dir * (2048u * 512u));
    }
    bias_cat<<<16, 256, 0, stream>>>(
        (const float*)d_in[17], (const float*)d_in[18], (const float*)d_in[19],
        (const float*)d_in[20], (const float*)d_in[21], (const float*)d_in[22],
        (const float*)d_in[23], (const float*)d_in[24], bcat);

    // 4) input projections: [16384,512] x [512,2048] per direction (bwd time-reversed)
    gemm_in<<<dim3(128, 16, 2), 256, 0, stream>>>(xb, Wpk, bcat, P);

    // 5) recurrent scan: 32 persistent WGs (16 H-chunks x 2 dirs)
    lstm_scan<<<dim3(16, 2), 128, 0, stream>>>(Upk, P, hbuf, out, cnt);
}
